// ResGCN_58128087384882
// MI455X (gfx1250) — compile-verified, matrix operand order checked by reading, not run on hardware
//
#include <hip/hip_runtime.h>
#include <hip/hip_bf16.h>

#define N_NODES 100000
#define N_EDGES 1600000
#define NFEAT   256
#define NHID    128
#define M_TILES (N_NODES / 16)   // 6250

typedef __attribute__((ext_vector_type(16))) _Float16 v16h;
typedef __attribute__((ext_vector_type(8)))  _Float16 v8h;
typedef __attribute__((ext_vector_type(4)))  _Float16 v4h;
typedef __attribute__((ext_vector_type(8)))  float    v8f;

// ---------- fragment loaders (CDNA5 16-bit A/B layout, ISA 7.12.2) ----------
// lane L: m/n = L&15, half = L>>4
// elements 0..7  = K  8*half + 0..7
// elements 8..15 = K 16+8*half + 0..7
__device__ __forceinline__ v16h frag_from_f32(const float* __restrict__ p, int half) {
    const float4 f0 = *(const float4*)(p + 8 * half);
    const float4 f1 = *(const float4*)(p + 8 * half + 4);
    const float4 f2 = *(const float4*)(p + 16 + 8 * half);
    const float4 f3 = *(const float4*)(p + 16 + 8 * half + 4);
    v16h r;
    r[0]  = (_Float16)f0.x; r[1]  = (_Float16)f0.y; r[2]  = (_Float16)f0.z; r[3]  = (_Float16)f0.w;
    r[4]  = (_Float16)f1.x; r[5]  = (_Float16)f1.y; r[6]  = (_Float16)f1.z; r[7]  = (_Float16)f1.w;
    r[8]  = (_Float16)f2.x; r[9]  = (_Float16)f2.y; r[10] = (_Float16)f2.z; r[11] = (_Float16)f2.w;
    r[12] = (_Float16)f3.x; r[13] = (_Float16)f3.y; r[14] = (_Float16)f3.z; r[15] = (_Float16)f3.w;
    return r;
}

__device__ __forceinline__ v16h frag_from_f16(const _Float16* __restrict__ p, int half) {
    v8h lo = *(const v8h*)(p + 8 * half);
    v8h hi = *(const v8h*)(p + 16 + 8 * half);
    v16h r;
#pragma unroll
    for (int i = 0; i < 8; ++i) { r[i] = lo[i]; r[8 + i] = hi[i]; }
    return r;
}

// ---------- weight prep: f32 [K][N] -> f16 transposed [n][k] ----------
__global__ void __launch_bounds__(256) prep_weights(const float* __restrict__ Wres,
                                                    const float* __restrict__ W1,
                                                    const float* __restrict__ W2,
                                                    _Float16* __restrict__ WresT,
                                                    _Float16* __restrict__ W1T,
                                                    _Float16* __restrict__ W2T) {
    int t = blockIdx.x * 256 + threadIdx.x;
    const int S1 = NFEAT * NHID;   // 32768
    const int S2 = NHID * NHID;    // 16384
    if (t < S1) {
        int k = t / NHID, n = t % NHID;
        WresT[n * NFEAT + k] = (_Float16)Wres[t];
    } else if (t < 2 * S1) {
        int u = t - S1;
        int k = u / NHID, n = u % NHID;
        W1T[n * NFEAT + k] = (_Float16)W1[u];
    } else if (t < 2 * S1 + S2) {
        int u = t - 2 * S1;
        int k = u / NHID, n = u % NHID;
        W2T[n * NHID + k] = (_Float16)W2[u];
    }
}

// ---------- GEMM1: z = x@W_res + b_res ; y1 = x@W1 ----------
// 8 waves: waves 0-3 -> W_res on m-tiles blockIdx*4+0..3, waves 4-7 -> W1 on same tiles.
// Each wave owns one m-tile and all 8 n-tiles: one A-fragment load feeds 8 WMMAs.
__global__ void __launch_bounds__(256) gemm1_kernel(const float* __restrict__ x,
                                                    const _Float16* __restrict__ WresT,
                                                    const _Float16* __restrict__ W1T,
                                                    const float* __restrict__ b_res,
                                                    float* __restrict__ z,
                                                    float* __restrict__ y1) {
    const int lane = threadIdx.x & 31;
    const int wave = threadIdx.x >> 5;
    const int mat  = wave >> 2;                 // 0: W_res -> z, 1: W1 -> y1
    int mt = blockIdx.x * 4 + (wave & 3);
    const bool active = (mt < M_TILES);
    if (mt >= M_TILES) mt = M_TILES - 1;        // clamp loads; stores guarded below
    const int half = lane >> 4;
    const int l16  = lane & 15;
    const int m0   = mt * 16;

    const float*    arow = x + (size_t)(m0 + l16) * NFEAT;
    const _Float16* Wt   = mat ? W1T : WresT;
    const _Float16* brow = Wt + (size_t)l16 * NFEAT;

    v8f acc[8];
#pragma unroll
    for (int n = 0; n < 8; ++n) acc[n] = (v8f){};

    for (int kc = 0; kc < NFEAT / 32; ++kc) {
        const int k0 = kc * 32;
        v16h a = frag_from_f32(arow + k0, half);
#pragma unroll
        for (int n = 0; n < 8; ++n) {
            v16h b = frag_from_f16(brow + (size_t)n * 16 * NFEAT + k0, half);
            acc[n] = __builtin_amdgcn_wmma_f32_16x16x32_f16(false, a, false, b, (short)0, acc[n], false, false);
        }
    }

    if (!active) return;                        // wave-uniform, after all WMMAs
    float* dst = mat ? y1 : z;
#pragma unroll
    for (int n = 0; n < 8; ++n) {
        const int col  = n * 16 + l16;
        const float bias = mat ? 0.0f : b_res[col];
#pragma unroll
        for (int d = 0; d < 8; ++d) {
            const int row = m0 + d + 8 * half;  // C/D layout: VGPR d -> M = d (+8 for hi half)
            dst[(size_t)row * NHID + col] = acc[n][d] + bias;
        }
    }
}

// ---------- GEMM2: y2 = x1h @ W2 (K = 128), one m-tile per wave, 8 n-tiles ----------
__global__ void __launch_bounds__(256) gemm2_kernel(const _Float16* __restrict__ x1h,
                                                    const _Float16* __restrict__ W2T,
                                                    float* __restrict__ y2) {
    const int lane = threadIdx.x & 31;
    const int wave = threadIdx.x >> 5;
    int mt = blockIdx.x * 8 + wave;
    const bool active = (mt < M_TILES);
    if (mt >= M_TILES) mt = M_TILES - 1;
    const int half = lane >> 4;
    const int l16  = lane & 15;
    const int m0   = mt * 16;

    const _Float16* arow = x1h + (size_t)(m0 + l16) * NHID;
    const _Float16* brow = W2T + (size_t)l16 * NHID;

    v8f acc[8];
#pragma unroll
    for (int n = 0; n < 8; ++n) acc[n] = (v8f){};

    for (int kc = 0; kc < NHID / 32; ++kc) {
        const int k0 = kc * 32;
        v16h a = frag_from_f16(arow + k0, half);
#pragma unroll
        for (int n = 0; n < 8; ++n) {
            v16h b = frag_from_f16(brow + (size_t)n * 16 * NHID + k0, half);
            acc[n] = __builtin_amdgcn_wmma_f32_16x16x32_f16(false, a, false, b, (short)0, acc[n], false, false);
        }
    }

    if (!active) return;
#pragma unroll
    for (int n = 0; n < 8; ++n) {
        const int col = n * 16 + l16;
#pragma unroll
        for (int d = 0; d < 8; ++d) {
            const int row = m0 + d + 8 * half;
            y2[(size_t)row * NHID + col] = acc[n][d];
        }
    }
}

// ---------- sorted-row CSR range via binary search ----------
__device__ __forceinline__ int lower_bound(const int* __restrict__ a, int n, int key) {
    int lo = 0, hi = n;
    while (lo < hi) {
        int mid = (lo + hi) >> 1;
        if (a[mid] < key) lo = mid + 1; else hi = mid;
    }
    return lo;
}

// ---------- SpMM1: x1 = relu(spmm(y1) + b1) + z  -> f16 ----------
// lane covers columns lane*4 .. lane*4+3 -> one b128 gather per edge per lane.
__global__ void __launch_bounds__(256) spmm1_kernel(const int* __restrict__ erow,
                                                    const int* __restrict__ ecol,
                                                    const float* __restrict__ eval,
                                                    const float* __restrict__ y1,
                                                    const float* __restrict__ b1,
                                                    const float* __restrict__ z,
                                                    _Float16* __restrict__ x1h) {
    const int lane = threadIdx.x & 31;
    const int node = blockIdx.x * 8 + (threadIdx.x >> 5);

    const int start = lower_bound(erow, N_EDGES, node);
    const int end   = lower_bound(erow, N_EDGES, node + 1);

    float4 acc = {0.f, 0.f, 0.f, 0.f};
    for (int e = start; e < end; ++e) {
        const int   c = ecol[e];
        const float v = eval[e];
        const float4 f = ((const float4*)(y1 + (size_t)c * NHID))[lane];
        acc.x = fmaf(v, f.x, acc.x);
        acc.y = fmaf(v, f.y, acc.y);
        acc.z = fmaf(v, f.z, acc.z);
        acc.w = fmaf(v, f.w, acc.w);
    }

    const float4 bv = ((const float4*)b1)[lane];
    const float4 zv = ((const float4*)(z + (size_t)node * NHID))[lane];
    v4h o;
    o[0] = (_Float16)(fmaxf(acc.x + bv.x, 0.0f) + zv.x);
    o[1] = (_Float16)(fmaxf(acc.y + bv.y, 0.0f) + zv.y);
    o[2] = (_Float16)(fmaxf(acc.z + bv.z, 0.0f) + zv.z);
    o[3] = (_Float16)(fmaxf(acc.w + bv.w, 0.0f) + zv.w);
    ((v4h*)(x1h + (size_t)node * NHID))[lane] = o;
}

// ---------- SpMM2 + bias + log_softmax ----------
__global__ void __launch_bounds__(256) spmm2_softmax_kernel(const int* __restrict__ erow,
                                                            const int* __restrict__ ecol,
                                                            const float* __restrict__ eval,
                                                            const float* __restrict__ y2,
                                                            const float* __restrict__ b2,
                                                            float* __restrict__ out) {
    const int lane = threadIdx.x & 31;
    const int node = blockIdx.x * 8 + (threadIdx.x >> 5);

    const int start = lower_bound(erow, N_EDGES, node);
    const int end   = lower_bound(erow, N_EDGES, node + 1);

    float4 acc = {0.f, 0.f, 0.f, 0.f};
    for (int e = start; e < end; ++e) {
        const int   c = ecol[e];
        const float v = eval[e];
        const float4 f = ((const float4*)(y2 + (size_t)c * NHID))[lane];
        acc.x = fmaf(v, f.x, acc.x);
        acc.y = fmaf(v, f.y, acc.y);
        acc.z = fmaf(v, f.z, acc.z);
        acc.w = fmaf(v, f.w, acc.w);
    }

    const float4 bv = ((const float4*)b2)[lane];
    float h[4] = {acc.x + bv.x, acc.y + bv.y, acc.z + bv.z, acc.w + bv.w};

    // wave32 log_softmax over 128 values (4 contiguous per lane)
    float m = fmaxf(fmaxf(h[0], h[1]), fmaxf(h[2], h[3]));
#pragma unroll
    for (int mask = 16; mask >= 1; mask >>= 1) m = fmaxf(m, __shfl_xor(m, mask, 32));

    float s = 0.f;
#pragma unroll
    for (int j = 0; j < 4; ++j) s += __expf(h[j] - m);
#pragma unroll
    for (int mask = 16; mask >= 1; mask >>= 1) s += __shfl_xor(s, mask, 32);

    const float ls = __logf(s);
    float4 o = {h[0] - m - ls, h[1] - m - ls, h[2] - m - ls, h[3] - m - ls};
    ((float4*)(out + (size_t)node * NHID))[lane] = o;
}

// ---------- workspace layout (bytes) ----------
#define WS_WRES_T   ((size_t)0)                      //  65,536  f16 [128][256]
#define WS_W1_T     ((size_t)65536)                  //  65,536  f16 [128][256]
#define WS_W2_T     ((size_t)131072)                 //  32,768  f16 [128][128]
#define WS_Z        ((size_t)163840)                 //  51.2 MB f32 [N][128]
#define WS_Y1       ((size_t)51363840)               //  51.2 MB f32 [N][128]
#define WS_X1H      ((size_t)102563840)              //  25.6 MB f16 [N][128]
#define WS_Y2       ((size_t)128163840)              //  51.2 MB f32 [N][128]

extern "C" void kernel_launch(void* const* d_in, const int* in_sizes, int n_in,
                              void* d_out, int out_size, void* d_ws, size_t ws_size,
                              hipStream_t stream) {
    const float* x     = (const float*)d_in[0];
    const int*   erow  = (const int*)  d_in[1];
    const int*   ecol  = (const int*)  d_in[2];
    const float* eval  = (const float*)d_in[3];
    const float* Wres  = (const float*)d_in[4];
    const float* bres  = (const float*)d_in[5];
    const float* W1    = (const float*)d_in[6];
    const float* b1    = (const float*)d_in[7];
    const float* W2    = (const float*)d_in[8];
    const float* b2    = (const float*)d_in[9];

    char* ws = (char*)d_ws;
    _Float16* WresT = (_Float16*)(ws + WS_WRES_T);
    _Float16* W1T   = (_Float16*)(ws + WS_W1_T);
    _Float16* W2T   = (_Float16*)(ws + WS_W2_T);
    float*    z     = (float*)   (ws + WS_Z);
    float*    y1    = (float*)   (ws + WS_Y1);
    _Float16* x1h   = (_Float16*)(ws + WS_X1H);
    float*    y2    = (float*)   (ws + WS_Y2);
    float*    out   = (float*)d_out;

    prep_weights<<<320, 256, 0, stream>>>(Wres, W1, W2, WresT, W1T, W2T);
    gemm1_kernel<<<(M_TILES + 3) / 4, 256, 0, stream>>>(x, WresT, W1T, bres, z, y1);
    spmm1_kernel<<<N_NODES / 8, 256, 0, stream>>>(erow, ecol, eval, y1, b1, z, x1h);
    gemm2_kernel<<<(M_TILES + 7) / 8, 256, 0, stream>>>(x1h, W2T, y2);
    spmm2_softmax_kernel<<<N_NODES / 8, 256, 0, stream>>>(erow, ecol, eval, y2, b2, out);
}
